// SARDecoder_67156108640626
// MI455X (gfx1250) — matrix operand order repaired and checked by reading, not compiled, not run on hardware
//
#include <hip/hip_runtime.h>
#include <hip/hip_bf16.h>
#include <math.h>

typedef __attribute__((ext_vector_type(16))) _Float16 v16h;
typedef __attribute__((ext_vector_type(8)))  _Float16 v8h;
typedef __attribute__((ext_vector_type(8)))  float    v8f;
typedef _Float16 half_t;

#define BB   32
#define CCH  512
#define HHH  6
#define WWW  40
#define HWP  240      // 6*40
#define NCLS 97
#define LLAB 26
#define TDEC 27

static inline int cdiv_i(long a, long b) { return (int)((a + b - 1) / b); }

// ---------------------------------------------------------------------------
// Generic WMMA GEMM:  D(M,N) f32 = A(M,K) f16 @ W(N,K)^T f16 [+ Cadd] [+ bias]
// Wave computes a 16x64 tile of D.
//   REMAP=1 : gather A rows as shifted conv pixels (SAME pad -> zero rows)
//   NGUARD=1: bounds-check M/N (only needed for the N=97 pred head)
// All 5 fragments of a k-step are loaded before the 4 WMMAs are issued so the
// loads pipeline in one clause and the WMMAs run back-to-back.
// ---------------------------------------------------------------------------
template<int REMAP, int NGUARD>
__global__ void __launch_bounds__(128)
gemm_wmma_kernel(const half_t* __restrict__ A, long lda,
                 const half_t* __restrict__ Bw, long ldb,
                 const float* __restrict__ Cadd, long ldcadd,
                 const float* __restrict__ bias,
                 float* __restrict__ D, long ldd,
                 int M, int N, int K, int dy, int dx)
{
    int wave   = blockIdx.x * (blockDim.x >> 5) + (threadIdx.x >> 5);
    int lane   = threadIdx.x & 31;
    int tilesN = (N + 63) >> 6;
    int tilesM = (M + 15) >> 4;
    if (wave >= tilesM * tilesN) return;   // whole-wave uniform exit
    int tm = wave / tilesN;
    int tn = wave % tilesN;
    int lane15 = lane & 15;
    int mbase  = tm * 16 + ((lane < 16) ? 0 : 8);
    int n0     = tn * 64;

    v8f acc[4];
#pragma unroll
    for (int j = 0; j < 4; ++j) {
        int col = n0 + j * 16 + lane15;
#pragma unroll
        for (int r = 0; r < 8; ++r) {
            int m = mbase + r;
            float v = 0.f;
            if (!NGUARD || (col < N && m < M)) {
                if (Cadd) v += Cadd[(long)m * ldcadd + col];
                if (bias) v += bias[col];
            }
            acc[j][r] = v;
        }
    }

    int  arow   = tm * 16 + lane15;
    bool avalid = true;
    long aoff   = (long)arow * lda;
    if (REMAP) {
        // arow enumerates output pixels over (B=32, H=6, W=40)
        int p = arow;
        int bb = p / HWP, rem = p % HWP;
        int h = rem / WWW, w = rem % WWW;
        int hs = h + dy, ws2 = w + dx;
        avalid = (hs >= 0) && (hs < HHH) && (ws2 >= 0) && (ws2 < WWW);
        aoff = (long)(bb * HWP + hs * WWW + ws2) * lda;
    } else if (NGUARD) {
        avalid = (arow < M);
    }
    int klo   = (lane < 16) ? 0 : 8;    // A: elems 0..7 -> K=klo..klo+7, 8..15 -> K=16+klo..
    int khalf = (lane < 16) ? 0 : 8;    // B: elems 0..7 -> K=khalf..khalf+7, 8..15 -> K=16+khalf..

    for (int kb = 0; kb < K; kb += 32) {
        // ---- load A fragment ----
        v16h afrag;
        if (REMAP || NGUARD) {
            if (avalid) {
                const half_t* ap = A + aoff + kb;
                v8h alo = *(const v8h*)(ap + klo);
                v8h ahi = *(const v8h*)(ap + 16 + klo);
#pragma unroll
                for (int i = 0; i < 8; ++i) { afrag[i] = alo[i]; afrag[8 + i] = ahi[i]; }
            } else {
#pragma unroll
                for (int i = 0; i < 16; ++i) afrag[i] = (half_t)0.f;
            }
        } else {
            const half_t* ap = A + aoff + kb;
            v8h alo = *(const v8h*)(ap + klo);
            v8h ahi = *(const v8h*)(ap + 16 + klo);
#pragma unroll
            for (int i = 0; i < 8; ++i) { afrag[i] = alo[i]; afrag[8 + i] = ahi[i]; }
        }
        // ---- load all 4 B fragments ----
        v16h bfrag[4];
#pragma unroll
        for (int j = 0; j < 4; ++j) {
            int n = n0 + j * 16 + lane15;
            if (!NGUARD || (n < N)) {
                const half_t* bp = Bw + (long)n * ldb + kb + khalf;
                v8h b0 = *(const v8h*)(bp);
                v8h b1 = *(const v8h*)(bp + 16);
#pragma unroll
                for (int i = 0; i < 8; ++i) { bfrag[j][i] = b0[i]; bfrag[j][8 + i] = b1[i]; }
            } else {
#pragma unroll
                for (int i = 0; i < 16; ++i) bfrag[j][i] = (half_t)0.f;
            }
        }
        // ---- 4 back-to-back WMMAs ----
#pragma unroll
        for (int j = 0; j < 4; ++j) {
            acc[j] = __builtin_amdgcn_wmma_f32_16x16x32_f16(
                false, afrag, false, bfrag[j], (short)0, acc[j], false, false);
        }
    }

#pragma unroll
    for (int j = 0; j < 4; ++j) {
        int col = n0 + j * 16 + lane15;
        if (NGUARD && col >= N) continue;
#pragma unroll
        for (int r = 0; r < 8; ++r) {
            int m = mbase + r;
            if (!NGUARD || m < M) D[(long)m * ldd + col] = acc[j][r];
        }
    }
}

// ---------------------------------------------------------------------------
// Elementwise / helper kernels
// ---------------------------------------------------------------------------
__global__ void fill_f32_kernel(float* p, int n) {
    int i = blockIdx.x * blockDim.x + threadIdx.x;
    if (i < n) p[i] = 0.f;
}
__global__ void fill_f16_kernel(half_t* p, int n) {
    int i = blockIdx.x * blockDim.x + threadIdx.x;
    if (i < n) p[i] = (half_t)0.f;
}
__global__ void cvt_f32_f16_kernel(const float* __restrict__ s, half_t* __restrict__ d, int n) {
    int i = blockIdx.x * blockDim.x + threadIdx.x;
    if (i < n) d[i] = (half_t)s[i];
}
// k_W (512,512,3,3) -> 9 tap matrices (512x512) f16, tap-major
__global__ void cvt_kw_kernel(const float* __restrict__ kw, half_t* __restrict__ ktap) {
    int i = blockIdx.x * blockDim.x + threadIdx.x;
    if (i >= 9 * CCH * CCH) return;
    int tap = i / (CCH * CCH);
    int rem = i % (CCH * CCH);
    ktap[i] = (half_t)kw[(long)rem * 9 + tap];
}
__global__ void bias_sum_kernel(const float* __restrict__ a, const float* __restrict__ b,
                                float* __restrict__ o, int n) {
    int i = blockIdx.x * blockDim.x + threadIdx.x;
    if (i < n) o[i] = a[i] + b[i];
}
// feat (B,C,H,W) -> featv (B,W,C) f16 (max over H)
__global__ void maxpool_kernel(const float* __restrict__ feat, half_t* __restrict__ featv) {
    int i = blockIdx.x * blockDim.x + threadIdx.x;
    if (i >= BB * WWW * CCH) return;
    int c = i % CCH;
    int w = (i / CCH) % WWW;
    int b = i / (CCH * WWW);
    float m = -INFINITY;
    for (int h = 0; h < HHH; ++h)
        m = fmaxf(m, feat[(((long)b * CCH + c) * HHH + h) * WWW + w]);
    featv[i] = (half_t)m;
}
// feat (B,C,H,W) -> featpix (B,HW,C) f16
__global__ void featpix_kernel(const float* __restrict__ feat, half_t* __restrict__ fp) {
    int i = blockIdx.x * blockDim.x + threadIdx.x;
    if (i >= BB * HWP * CCH) return;
    int c  = i % CCH;
    int hw = (i / CCH) % HWP;
    int b  = i / (CCH * HWP);
    fp[i] = (half_t)feat[((long)b * CCH + c) * HWP + hw];
}
// LSTM cell: G (32,2048) pre-activations -> update c, write h (f16 buf + f16 seq)
__global__ void lstm_cell_kernel(const float* __restrict__ G,
                                 float* __restrict__ cst,
                                 half_t* __restrict__ hbuf,
                                 half_t* __restrict__ hseq, long rs)
{
    int idx = blockIdx.x * blockDim.x + threadIdx.x;
    if (idx >= BB * CCH) return;
    int m = idx >> 9, n = idx & 511;
    const float* g = G + (long)m * 2048;
    float gi = g[n], gf = g[512 + n], gg = g[1024 + n], go = g[1536 + n];
    float si = 1.f / (1.f + expf(-gi));
    float sf = 1.f / (1.f + expf(-gf));
    float so = 1.f / (1.f + expf(-go));
    float cc = sf * cst[idx] + si * tanhf(gg);
    float hh = so * tanhf(cc);
    cst[idx]  = cc;
    hbuf[idx] = (half_t)hh;
    hseq[(long)m * rs + n] = (half_t)hh;
}
// holistic f32 (32,512) -> tok row t=0 (f16) and concat[:,1024:1536) for all t
__global__ void holistic_spread_kernel(const float* __restrict__ hol,
                                       half_t* __restrict__ tok,
                                       half_t* __restrict__ cc)
{
    int idx = blockIdx.x * blockDim.x + threadIdx.x;
    if (idx >= BB * CCH) return;
    int m = idx >> 9, n = idx & 511;
    half_t hf = (half_t)hol[idx];
    tok[((long)m * TDEC) * CCH + n] = hf;
    for (int t = 0; t < TDEC; ++t)
        cc[((long)(m * TDEC + t)) * 1536 + 1024 + n] = hf;
}
// tok rows t=1..26 from embedding table
__global__ void tok_embed_kernel(const int* __restrict__ label, const float* __restrict__ emb,
                                 half_t* __restrict__ tok)
{
    int idx = blockIdx.x * blockDim.x + threadIdx.x;
    if (idx >= BB * LLAB * CCH) return;
    int n  = idx % CCH;
    int t1 = (idx / CCH) % LLAB;
    int m  = idx / (CCH * LLAB);
    int lbl = label[m * LLAB + t1];
    tok[((long)(m * TDEC + t1 + 1)) * CCH + n] = (half_t)emb[(long)lbl * CCH + n];
}
// score[b,t,hw] = sum_c tanh(k[b,hw,c] + q[b,t,c]) * sW[c] + sb ; block per (b,hw)
__global__ void score_kernel(const float* __restrict__ kout, const float* __restrict__ q,
                             const float* __restrict__ sW, const float* __restrict__ sb,
                             float* __restrict__ score)
{
    int bp = blockIdx.x;             // b*240 + pix
    int b  = bp / HWP;
    __shared__ float kr[CCH];
    __shared__ float sw[CCH];
    for (int c = threadIdx.x; c < CCH; c += blockDim.x) {
        kr[c] = kout[(long)bp * CCH + c];
        sw[c] = sW[c];
    }
    __syncthreads();
    int wid = threadIdx.x >> 5, lane = threadIdx.x & 31;
    int nw  = blockDim.x >> 5;
    float sbv = sb[0];
    for (int t = wid; t < TDEC; t += nw) {
        const float* qr = q + (long)(b * TDEC + t) * CCH;
        float part = 0.f;
        for (int c = lane; c < CCH; c += 32)
            part += tanhf(kr[c] + qr[c]) * sw[c];
        for (int off = 16; off > 0; off >>= 1) part += __shfl_down(part, off, 32);
        if (lane == 0) score[(long)(b * TDEC + t) * HWP + (bp % HWP)] = part + sbv;
    }
}
// masked softmax over 240 positions; block per (b,t)
__global__ void softmax_kernel(const float* __restrict__ score, const float* __restrict__ vr,
                               float* __restrict__ aw)
{
    int bt = blockIdx.x;
    int b  = bt / TDEC;
    __shared__ float sh[256];
    int tid = threadIdx.x;
    int vw  = (int)ceilf((float)WWW * vr[b]);
    if (vw > WWW) vw = WWW;
    float v = -INFINITY;
    if (tid < HWP) {
        int w = tid % WWW;
        float s = score[(long)bt * HWP + tid];
        v = (w >= vw) ? -INFINITY : s;
    }
    sh[tid] = v; __syncthreads();
    for (int s2 = 128; s2 > 0; s2 >>= 1) { if (tid < s2) sh[tid] = fmaxf(sh[tid], sh[tid + s2]); __syncthreads(); }
    float mx = sh[0]; __syncthreads();
    float e = (tid < HWP && v != -INFINITY) ? expf(v - mx) : 0.f;
    sh[tid] = e; __syncthreads();
    for (int s2 = 128; s2 > 0; s2 >>= 1) { if (tid < s2) sh[tid] += sh[tid + s2]; __syncthreads(); }
    float inv = 1.f / sh[0];
    if (tid < HWP) aw[(long)bt * HWP + tid] = e * inv;
}
// attn_feat[b,t,c] = sum_hw feat[b,c,hw]*aw[b,t,hw]  -> concat[:,512:1024) f16
__global__ void attn_kernel(const float* __restrict__ feat, const float* __restrict__ aw,
                            half_t* __restrict__ cc)
{
    int bt = blockIdx.x;
    int b  = bt / TDEC;
    __shared__ float a_s[HWP];
    for (int i = threadIdx.x; i < HWP; i += blockDim.x) a_s[i] = aw[(long)bt * HWP + i];
    __syncthreads();
    for (int c = threadIdx.x; c < CCH; c += blockDim.x) {
        const float* f = feat + ((long)b * CCH + c) * HWP;
        float s = 0.f;
        for (int hw = 0; hw < HWP; ++hw) s += f[hw] * a_s[hw];
        cc[(long)bt * 1536 + 512 + c] = (half_t)s;
    }
}
// d_out = preds[:,1:,:]
__global__ void out_gather_kernel(const float* __restrict__ preds, float* __restrict__ out) {
    int idx = blockIdx.x * blockDim.x + threadIdx.x;
    if (idx >= BB * LLAB * NCLS) return;
    int cl = idx % NCLS;
    int t1 = (idx / NCLS) % LLAB;
    int b  = idx / (NCLS * LLAB);
    out[idx] = preds[(long)(b * TDEC + t1 + 1) * NCLS + cl];
}

// ---------------------------------------------------------------------------
// Host side
// ---------------------------------------------------------------------------
static void launch_gemm(hipStream_t s, const half_t* A, long lda, const half_t* Bw, long ldb,
                        const float* Cadd, long ldcadd, const float* bias,
                        float* D, long ldd, int M, int N, int K)
{
    int tiles  = ((M + 15) / 16) * ((N + 63) / 64);
    int blocks = (tiles + 3) / 4;
    bool full  = ((M & 15) == 0) && ((N & 63) == 0);
    if (full)
        gemm_wmma_kernel<0, 0><<<blocks, 128, 0, s>>>(A, lda, Bw, ldb, Cadd, ldcadd, bias,
                                                      D, ldd, M, N, K, 0, 0);
    else
        gemm_wmma_kernel<0, 1><<<blocks, 128, 0, s>>>(A, lda, Bw, ldb, Cadd, ldcadd, bias,
                                                      D, ldd, M, N, K, 0, 0);
}
static void launch_gemm_tap(hipStream_t s, const half_t* A, long lda, const half_t* Bw, long ldb,
                            const float* Cadd, long ldcadd, const float* bias,
                            float* D, long ldd, int M, int N, int K, int dy, int dx)
{
    int tiles  = ((M + 15) / 16) * ((N + 63) / 64);
    int blocks = (tiles + 3) / 4;
    gemm_wmma_kernel<1, 0><<<blocks, 128, 0, s>>>(A, lda, Bw, ldb, Cadd, ldcadd, bias,
                                                  D, ldd, M, N, K, dy, dx);
}

extern "C" void kernel_launch(void* const* d_in, const int* in_sizes, int n_in,
                              void* d_out, int out_size, void* d_ws, size_t ws_size,
                              hipStream_t stream)
{
    (void)in_sizes; (void)n_in; (void)out_size; (void)ws_size;
    const float* feat   = (const float*)d_in[0];
    const int*   label  = (const int*)  d_in[1];
    const float* vr     = (const float*)d_in[2];
    const float* e0_Wih = (const float*)d_in[3],  *e0_Whh = (const float*)d_in[4];
    const float* e0_bih = (const float*)d_in[5],  *e0_bhh = (const float*)d_in[6];
    const float* e1_Wih = (const float*)d_in[7],  *e1_Whh = (const float*)d_in[8];
    const float* e1_bih = (const float*)d_in[9],  *e1_bhh = (const float*)d_in[10];
    const float* enc_W  = (const float*)d_in[11], *enc_b  = (const float*)d_in[12];
    const float* q_W    = (const float*)d_in[13], *q_b    = (const float*)d_in[14];
    const float* k_W    = (const float*)d_in[15], *k_b    = (const float*)d_in[16];
    const float* s_W    = (const float*)d_in[17], *s_b    = (const float*)d_in[18];
    const float* emb    = (const float*)d_in[19];
    const float* d0_Wih = (const float*)d_in[20], *d0_Whh = (const float*)d_in[21];
    const float* d0_bih = (const float*)d_in[22], *d0_bhh = (const float*)d_in[23];
    const float* d1_Wih = (const float*)d_in[24], *d1_Whh = (const float*)d_in[25];
    const float* d1_bih = (const float*)d_in[26], *d1_bhh = (const float*)d_in[27];
    const float* pred_W = (const float*)d_in[28], *pred_b = (const float*)d_in[29];
    float* out = (float*)d_out;

    // workspace carving (256B aligned)
    char* base = (char*)d_ws; size_t off = 0;
    auto alloc = [&](size_t bytes) -> void* {
        void* p = base + off; off = (off + bytes + 255) & ~(size_t)255; return p;
    };
    const size_t WSZ = 2048 * 512;
    half_t* e0WihH = (half_t*)alloc(WSZ * 2); half_t* e0WhhH = (half_t*)alloc(WSZ * 2);
    half_t* e1WihH = (half_t*)alloc(WSZ * 2); half_t* e1WhhH = (half_t*)alloc(WSZ * 2);
    half_t* d0WihH = (half_t*)alloc(WSZ * 2); half_t* d0WhhH = (half_t*)alloc(WSZ * 2);
    half_t* d1WihH = (half_t*)alloc(WSZ * 2); half_t* d1WhhH = (half_t*)alloc(WSZ * 2);
    half_t* encWH  = (half_t*)alloc(512 * 512 * 2);
    half_t* qWH    = (half_t*)alloc(512 * 512 * 2);
    half_t* predWH = (half_t*)alloc((size_t)NCLS * 1536 * 2);
    half_t* ktapH  = (half_t*)alloc((size_t)9 * 512 * 512 * 2);
    float* bsum_e0 = (float*)alloc(2048 * 4); float* bsum_e1 = (float*)alloc(2048 * 4);
    float* bsum_d0 = (float*)alloc(2048 * 4); float* bsum_d1 = (float*)alloc(2048 * 4);
    half_t* featv  = (half_t*)alloc((size_t)BB * WWW * CCH * 2);
    half_t* fpix   = (half_t*)alloc((size_t)BB * HWP * CCH * 2);
    float*  Xpre   = (float*)alloc((size_t)BB * 40 * 2048 * 4);
    float*  Gbuf   = (float*)alloc((size_t)BB * 2048 * 4);
    half_t* hbuf   = (half_t*)alloc((size_t)BB * CCH * 2);
    float*  cbuf   = (float*)alloc((size_t)BB * CCH * 4);
    half_t* hseq0  = (half_t*)alloc((size_t)BB * 40 * CCH * 2);
    half_t* hseq1  = (half_t*)alloc((size_t)BB * 40 * CCH * 2);
    half_t* dseq0  = (half_t*)alloc((size_t)BB * TDEC * CCH * 2);
    half_t* tok    = (half_t*)alloc((size_t)BB * TDEC * CCH * 2);
    half_t* concat = (half_t*)alloc((size_t)BB * TDEC * 1536 * 2);
    float*  holst  = (float*)alloc((size_t)BB * CCH * 4);
    float*  qbuf   = (float*)alloc((size_t)BB * TDEC * CCH * 4);
    float*  kout   = (float*)alloc((size_t)BB * HWP * CCH * 4);
    float*  score  = (float*)alloc((size_t)BB * TDEC * HWP * 4);
    float*  awbuf  = (float*)alloc((size_t)BB * TDEC * HWP * 4);
    float*  preds  = (float*)alloc((size_t)BB * TDEC * NCLS * 4);

    auto cvt = [&](const float* s, half_t* d, int n) {
        cvt_f32_f16_kernel<<<cdiv_i(n, 256), 256, 0, stream>>>(s, d, n);
    };
    // weight conversions
    cvt(e0_Wih, e0WihH, (int)WSZ); cvt(e0_Whh, e0WhhH, (int)WSZ);
    cvt(e1_Wih, e1WihH, (int)WSZ); cvt(e1_Whh, e1WhhH, (int)WSZ);
    cvt(d0_Wih, d0WihH, (int)WSZ); cvt(d0_Whh, d0WhhH, (int)WSZ);
    cvt(d1_Wih, d1WihH, (int)WSZ); cvt(d1_Whh, d1WhhH, (int)WSZ);
    cvt(enc_W, encWH, 512 * 512);
    cvt(q_W,   qWH,   512 * 512);
    cvt(pred_W, predWH, NCLS * 1536);
    cvt_kw_kernel<<<cdiv_i(9 * 512 * 512, 256), 256, 0, stream>>>(k_W, ktapH);
    bias_sum_kernel<<<8, 256, 0, stream>>>(e0_bih, e0_bhh, bsum_e0, 2048);
    bias_sum_kernel<<<8, 256, 0, stream>>>(e1_bih, e1_bhh, bsum_e1, 2048);
    bias_sum_kernel<<<8, 256, 0, stream>>>(d0_bih, d0_bhh, bsum_d0, 2048);
    bias_sum_kernel<<<8, 256, 0, stream>>>(d1_bih, d1_bhh, bsum_d1, 2048);

    // feature prep
    maxpool_kernel<<<cdiv_i(BB * WWW * CCH, 256), 256, 0, stream>>>(feat, featv);
    featpix_kernel<<<cdiv_i(BB * HWP * CCH, 256), 256, 0, stream>>>(feat, fpix);

    // conv3x3 'SAME' as 9 implicit-GEMM taps -> kout (B*HW, C) f32
    for (int tap = 0; tap < 9; ++tap) {
        int ky = tap / 3, kx = tap % 3;
        launch_gemm_tap(stream, fpix, 512, ktapH + (size_t)tap * 512 * 512, 512,
                        (tap ? kout : nullptr), 512, (tap == 0 ? k_b : nullptr),
                        kout, 512, BB * HWP, 512, 512, ky - 1, kx - 1);
    }

    // one LSTM layer: X (B*T,512) f16 -> hseq (rows (m*T+t), stride outLd)
    auto run_lstm_layer = [&](const half_t* X, const half_t* Wih, const half_t* Whh,
                              const float* bsum, int T, half_t* hseq, long outLd) {
        fill_f16_kernel<<<64, 256, 0, stream>>>(hbuf, BB * CCH);
        fill_f32_kernel<<<64, 256, 0, stream>>>(cbuf, BB * CCH);
        launch_gemm(stream, X, 512, Wih, 512, nullptr, 0, bsum, Xpre, 2048, BB * T, 2048, 512);
        for (int t = 0; t < T; ++t) {
            launch_gemm(stream, hbuf, 512, Whh, 512, Xpre + (long)t * 2048, (long)T * 2048,
                        nullptr, Gbuf, 2048, BB, 2048, 512);
            lstm_cell_kernel<<<64, 256, 0, stream>>>(Gbuf, cbuf, hbuf,
                                                     hseq + (long)t * outLd, (long)T * outLd);
        }
    };

    // encoder
    run_lstm_layer(featv, e0WihH, e0WhhH, bsum_e0, 40, hseq0, 512);
    run_lstm_layer(hseq0, e1WihH, e1WhhH, bsum_e1, 40, hseq1, 512);
    launch_gemm(stream, hseq1 + 39 * 512, 40 * 512, encWH, 512, nullptr, 0, enc_b,
                holst, 512, BB, 512, 512);
    holistic_spread_kernel<<<64, 256, 0, stream>>>(holst, tok, concat);
    tok_embed_kernel<<<cdiv_i(BB * LLAB * CCH, 256), 256, 0, stream>>>(label, emb, tok);

    // decoder (layer1 writes Hid directly into concat[:,0:512))
    run_lstm_layer(tok,   d0WihH, d0WhhH, bsum_d0, TDEC, dseq0, 512);
    run_lstm_layer(dseq0, d1WihH, d1WhhH, bsum_d1, TDEC, concat, 1536);

    // q = Hid @ qW^T + qb   (A = concat[:,0:512), lda=1536)
    launch_gemm(stream, concat, 1536, qWH, 512, nullptr, 0, q_b, qbuf, 512,
                BB * TDEC, 512, 512);

    // attention
    score_kernel<<<BB * HWP, 256, 0, stream>>>(kout, qbuf, s_W, s_b, score);
    softmax_kernel<<<BB * TDEC, 256, 0, stream>>>(score, vr, awbuf);
    attn_kernel<<<BB * TDEC, 256, 0, stream>>>(feat, awbuf, concat);

    // prediction head + output gather
    launch_gemm(stream, concat, 1536, predWH, 1536, nullptr, 0, pred_b, preds, NCLS,
                BB * TDEC, NCLS, 1536);
    out_gather_kernel<<<cdiv_i(BB * LLAB * NCLS, 256), 256, 0, stream>>>(preds, out);
}
